// Attention_70437463654670
// MI455X (gfx1250) — compile-verified, matrix operand order checked by reading, not run on hardware
//
#include <hip/hip_runtime.h>
#include <hip/hip_bf16.h>

#define DIMD 1024
#define HEADS 16
#define DIM_HEAD 64
#define BATCH 2
#define SEQ 2048
#define BN (BATCH * SEQ)             // 4096 rows
#define QKV_N (3 * HEADS * DIM_HEAD) // 3072

#define TDM_GEMM 1  // GEMM tiles via tensor_load_to_lds (TDM)
#define TDM_ATTN 1  // attention K/V tiles via tensor_load_to_lds (TDM)

typedef __attribute__((ext_vector_type(16))) __bf16 v16bf;
typedef __attribute__((ext_vector_type(8)))  __bf16 bf16x8;
typedef __attribute__((ext_vector_type(8)))  float  v8f;
typedef __attribute__((ext_vector_type(4)))  unsigned int u32x4;
typedef __attribute__((ext_vector_type(8)))  unsigned int u32x8;

union FragAB { v16bf v; bf16x8 h[2]; };

static __device__ __forceinline__ v8f wmma_bf16(v16bf a, v16bf b, v8f c) {
  return __builtin_amdgcn_wmma_f32_16x16x32_bf16(false, a, false, b, (short)0, c,
                                                 false, false);
}

#if TDM_GEMM || TDM_ATTN
// Issue one TDM 2-D tile load: global (row-major, stride0 elements) -> LDS
// (contiguous). D# per CDNA5 ISA: group0 =
// flags/lds/global/type, group1 = data_size, tensor dims, tile dims, dim0
// stride. Groups 2/3 zero (2-D tensor). Tracked by TENSORcnt.
static __device__ __forceinline__ void tdm_load_2d(unsigned int lds_off,
                                                   const __bf16* gptr,
                                                   unsigned int tensor_d0,
                                                   unsigned int tensor_d1,
                                                   unsigned long long stride0,
                                                   unsigned int tile_d0,
                                                   unsigned int tile_d1) {
  unsigned long long ga = (unsigned long long)(uintptr_t)gptr;
  u32x4 g0;
  g0[0] = 1u;                                      // count=1 (valid user D#)
  g0[1] = lds_off;                                 // lds_addr (bytes)
  g0[2] = (unsigned int)ga;                        // global_addr[31:0]
  g0[3] = ((unsigned int)(ga >> 32) & 0x01FFFFFFu) // global_addr[56:32]
          | (2u << 30);                            // type = 2 ("image")
  u32x8 g1;
  g1[0] = 1u << 16;                                // data_size = 1 -> 2 bytes
  g1[1] = (tensor_d0 & 0xFFFFu) << 16;             // tensor_dim0[15:0]
  g1[2] = (tensor_d0 >> 16) | ((tensor_d1 & 0xFFFFu) << 16);
  g1[3] = (tensor_d1 >> 16) | (tile_d0 << 16);     // tile_dim0
  g1[4] = tile_d1 & 0xFFFFu;                       // tile_dim1 (tile_dim2=0)
  g1[5] = (unsigned int)stride0;                   // tensor_dim0_stride[31:0]
  g1[6] = (unsigned int)(stride0 >> 32) & 0xFFFFu; // stride[47:32]; dim1_stride=0
  g1[7] = 0u;
  u32x4 gz;
  gz[0] = 0u; gz[1] = 0u; gz[2] = 0u; gz[3] = 0u;
  asm volatile("tensor_load_to_lds %0, %1, %2, %3"
               :
               : "s"(g0), "s"(g1), "s"(gz), "s"(gz)
               : "memory");
}
#endif

// ------------------------------------------------- weight convert+transpose
// w: K x N (f32, row-major) -> wT: N x K (bf16)
__global__ void cvt_transpose_bf16(const float* __restrict__ w, __bf16* __restrict__ wT,
                                   int K, int N) {
  int idx = blockIdx.x * blockDim.x + threadIdx.x;
  int kk = idx / N, nn = idx % N;
  if (kk < K) wT[(size_t)nn * K + kk] = (__bf16)w[idx];
}

// ------------------------------------------- RMSNorm (L2 variant) + gates
__global__ __launch_bounds__(256)
void rmsnorm_gates(const float* __restrict__ x, const float* __restrict__ gamma,
                   const float* __restrict__ w_gates, const float* __restrict__ b_gates,
                   __bf16* __restrict__ xn, float* __restrict__ gates) {
  __shared__ float red[256];
  __shared__ float xrow[DIMD];
  __shared__ float gp[HEADS][17];
  const int r = blockIdx.x;
  const int t = threadIdx.x;

  float4 xv = ((const float4*)(x + (size_t)r * DIMD))[t];
  float ss = xv.x * xv.x + xv.y * xv.y + xv.z * xv.z + xv.w * xv.w;
  red[t] = ss;
  __syncthreads();
  for (int s2 = 128; s2 > 0; s2 >>= 1) {
    if (t < s2) red[t] += red[t + s2];
    __syncthreads();
  }
  const float inv = rsqrtf(red[0]) * 32.0f;  // sqrt(1024)

  float4 gv = ((const float4*)gamma)[t];
  float o0 = xv.x * inv * gv.x;
  float o1 = xv.y * inv * gv.y;
  float o2 = xv.z * inv * gv.z;
  float o3 = xv.w * inv * gv.w;
  size_t base = (size_t)r * DIMD + t * 4;
  xn[base + 0] = (__bf16)o0; xn[base + 1] = (__bf16)o1;
  xn[base + 2] = (__bf16)o2; xn[base + 3] = (__bf16)o3;
  xrow[t * 4 + 0] = o0; xrow[t * 4 + 1] = o1;
  xrow[t * 4 + 2] = o2; xrow[t * 4 + 3] = o3;
  __syncthreads();

  const int h = t >> 4, l16 = t & 15;
  float p = 0.f;
  #pragma unroll 8
  for (int d0 = 0; d0 < 64; ++d0) {
    int d = l16 * 64 + d0;
    p += xrow[d] * w_gates[d * HEADS + h];
  }
  gp[h][l16] = p;
  __syncthreads();
  if (t < HEADS) {
    float sum = b_gates[t];
    #pragma unroll
    for (int i = 0; i < 16; ++i) sum += gp[t][i];
    gates[(size_t)r * HEADS + t] = 1.0f / (1.0f + __expf(-sum));
  }
}

// ------------------------------------------------ generic bf16 WMMA GEMM
// C(MxN) = A(MxK,row) @ BT(NxK,row)^T.  Block tile 128x128, 8 waves (4Mx2N),
// wave tile 32x64 = 8 accumulators, 8 WMMAs per K=32 step, double-buffered
// LDS tiles fed by the Tensor Data Mover.
template <typename TOut>
__global__ __launch_bounds__(256)
void gemm_bf16_wmma(const __bf16* __restrict__ A, const __bf16* __restrict__ BT,
                    TOut* __restrict__ C, int M, int N, int K) {
  __shared__ alignas(16) __bf16 ldsA[2][128][32];
  __shared__ alignas(16) __bf16 ldsB[2][128][32];  // BT rows: [n][k]
  const int t = threadIdx.x;
  const int lane = t & 31, wave = t >> 5;
  const int wm = wave & 3, wn = wave >> 2;
  const int m0 = blockIdx.y * 128, n0 = blockIdx.x * 128;
  const int rL = lane & 15, kh = (lane >> 4) * 8, hi8 = (lane >> 4) * 8;

  v8f acc[2][4];
  #pragma unroll
  for (int a = 0; a < 2; ++a)
    #pragma unroll
    for (int j = 0; j < 4; ++j)
      #pragma unroll
      for (int e = 0; e < 8; ++e) acc[a][j][e] = 0.0f;

  const int NKIT = K / 32;
#if TDM_GEMM
  const unsigned int ldsA_off = (unsigned int)(uintptr_t)&ldsA[0][0][0];
  const unsigned int ldsB_off = (unsigned int)(uintptr_t)&ldsB[0][0][0];
  const unsigned int bufBytes = (unsigned int)sizeof(ldsA[0]);
  if (wave == 0) {
    tdm_load_2d(ldsA_off, A + (size_t)m0 * K, (unsigned)K, (unsigned)M,
                (unsigned long long)K, 32u, 128u);
    tdm_load_2d(ldsB_off, BT + (size_t)n0 * K, (unsigned)K, (unsigned)N,
                (unsigned long long)K, 32u, 128u);
  }
#endif
  for (int it = 0; it < NKIT; ++it) {
    const int k0 = it * 32;
    const int cur = it & 1;
#if TDM_GEMM
    if (wave == 0) {
      if (it + 1 < NKIT) {  // prefetch next pair into the other buffer
        tdm_load_2d(ldsA_off + (cur ^ 1) * bufBytes,
                    A + (size_t)m0 * K + k0 + 32, (unsigned)K, (unsigned)M,
                    (unsigned long long)K, 32u, 128u);
        tdm_load_2d(ldsB_off + (cur ^ 1) * bufBytes,
                    BT + (size_t)n0 * K + k0 + 32, (unsigned)K, (unsigned)N,
                    (unsigned long long)K, 32u, 128u);
        __builtin_amdgcn_s_wait_tensorcnt((short)2);  // current pair done
      } else {
        __builtin_amdgcn_s_wait_tensorcnt((short)0);
      }
    }
#else
    {  // cooperative vector-load fallback
      int row = t >> 1, col = (t & 1) * 16;
      const __bf16* ap = &A[(size_t)(m0 + row) * K + k0 + col];
      *(bf16x8*)&ldsA[cur][row][col]     = *(const bf16x8*)ap;
      *(bf16x8*)&ldsA[cur][row][col + 8] = *(const bf16x8*)(ap + 8);
      const __bf16* bp = &BT[(size_t)(n0 + row) * K + k0 + col];
      *(bf16x8*)&ldsB[cur][row][col]     = *(const bf16x8*)bp;
      *(bf16x8*)&ldsB[cur][row][col + 8] = *(const bf16x8*)(bp + 8);
    }
#endif
    __syncthreads();
    FragAB af[2];
    #pragma unroll
    for (int a = 0; a < 2; ++a) {
      af[a].h[0] = *(const bf16x8*)&ldsA[cur][wm * 32 + a * 16 + rL][kh];
      af[a].h[1] = *(const bf16x8*)&ldsA[cur][wm * 32 + a * 16 + rL][16 + kh];
    }
    #pragma unroll
    for (int j = 0; j < 4; ++j) {
      FragAB bfr;
      bfr.h[0] = *(const bf16x8*)&ldsB[cur][wn * 64 + j * 16 + rL][kh];
      bfr.h[1] = *(const bf16x8*)&ldsB[cur][wn * 64 + j * 16 + rL][16 + kh];
      acc[0][j] = wmma_bf16(af[0].v, bfr.v, acc[0][j]);
      acc[1][j] = wmma_bf16(af[1].v, bfr.v, acc[1][j]);
    }
    __syncthreads();
  }
  #pragma unroll
  for (int a = 0; a < 2; ++a)
    #pragma unroll
    for (int j = 0; j < 4; ++j) {
      int col = n0 + wn * 64 + j * 16 + rL;
      #pragma unroll
      for (int i = 0; i < 8; ++i) {
        int row = m0 + wm * 32 + a * 16 + i + hi8;
        C[(size_t)row * N + col] = (TOut)acc[a][j][i];
      }
    }
}

// ---------------------------------------- RoPE (interleaved) + qkv scatter
// q,k -> [bh][n][dh];  v -> TRANSPOSED [bh][dh][n] so attention tiles are
// plain contiguous 2-D tiles (TDM-shaped, no LDS-side transpose).
__global__ void rope_scatter(const __bf16* __restrict__ qkv, const float* __restrict__ freqs,
                             __bf16* __restrict__ q, __bf16* __restrict__ k,
                             __bf16* __restrict__ v) {
  int idx = blockIdx.x * blockDim.x + threadIdx.x;  // BN*HEADS*32
  int u = idx & 31;
  int h = (idx >> 5) & (HEADS - 1);
  int r = idx >> 9;
  int bIdx = r / SEQ, n = r % SEQ;
  int bh = bIdx * HEADS + h;
  size_t base = (size_t)r * QKV_N + h * DIM_HEAD + 2 * u;
  float q0 = (float)qkv[base], q1 = (float)qkv[base + 1];
  float k0 = (float)qkv[base + 1024], k1 = (float)qkv[base + 1025];
  float v0 = (float)qkv[base + 2048], v1 = (float)qkv[base + 2049];
  float f = (float)n * freqs[u];
  float cf = __cosf(f), sf = __sinf(f);
  size_t dst = ((size_t)bh * SEQ + n) * DIM_HEAD + 2 * u;
  q[dst]     = (__bf16)(q0 * cf - q1 * sf);
  q[dst + 1] = (__bf16)(q1 * cf + q0 * sf);
  k[dst]     = (__bf16)(k0 * cf - k1 * sf);
  k[dst + 1] = (__bf16)(k1 * cf + k0 * sf);
  size_t dstv = ((size_t)bh * DIM_HEAD + 2 * u) * SEQ + n;
  v[dstv]       = (__bf16)v0;
  v[dstv + SEQ] = (__bf16)v1;
}

// ------------------------------------- flash attention + sigmoid gating
// grid: (SEQ/64, BATCH*HEADS); block: 128 (4 waves, 16 q-rows each).
// K/V tiles double-buffered in LDS, fed by the Tensor Data Mover so the DMA
// of block kb+32 overlaps WMMA+softmax of block kb.
__global__ __launch_bounds__(128)
void flash_attn_gated(const __bf16* __restrict__ q, const __bf16* __restrict__ k,
                      const __bf16* __restrict__ vT, const float* __restrict__ gates,
                      __bf16* __restrict__ out) {
  __shared__ alignas(16) __bf16 ldsK[2][32][64];
  __shared__ alignas(16) __bf16 ldsVT[2][64][32];   // [dh][k] straight from vT
  __shared__ alignas(16) __bf16 ldsP[4][16][32];    // per-wave P staging
  const int t = threadIdx.x, lane = t & 31, wave = t >> 5;
  const int bh = blockIdx.y, bIdx = bh / HEADS, h = bh % HEADS;
  const int q0row = blockIdx.x * 64 + wave * 16;
  const int rL = lane & 15, kh = (lane >> 4) * 8, hi8 = (lane >> 4) * 8;
  const size_t qb = (size_t)bh * SEQ * DIM_HEAD;
  const __bf16* kBase = k + qb;
  const __bf16* vBase = vT + (size_t)bh * DIM_HEAD * SEQ;

  FragAB qa[2];
  #pragma unroll
  for (int c = 0; c < 2; ++c) {
    const __bf16* qp = q + qb + (size_t)(q0row + rL) * DIM_HEAD + c * 32;
    qa[c].h[0] = *(const bf16x8*)(qp + kh);
    qa[c].h[1] = *(const bf16x8*)(qp + 16 + kh);
  }

  float mrow[8], lrow[8];
  v8f o[4];
  #pragma unroll
  for (int i = 0; i < 8; ++i) { mrow[i] = -1e30f; lrow[i] = 0.f; }
  #pragma unroll
  for (int j = 0; j < 4; ++j)
    #pragma unroll
    for (int e = 0; e < 8; ++e) o[j][e] = 0.f;

  const float scale = 0.125f;  // 1/sqrt(64)
  const int NKB = SEQ / 32;

#if TDM_ATTN
  const unsigned int ldsK_off = (unsigned int)(uintptr_t)&ldsK[0][0][0];
  const unsigned int ldsV_off = (unsigned int)(uintptr_t)&ldsVT[0][0][0];
  const unsigned int kBytes = (unsigned int)sizeof(ldsK[0]);
  const unsigned int vBytes = (unsigned int)sizeof(ldsVT[0]);
  if (wave == 0) {
    // K tile: 32 rows x 64 (contiguous rows, stride 64); V tile: 64 x 32, stride SEQ
    tdm_load_2d(ldsK_off, kBase, 64u, (unsigned)SEQ,
                (unsigned long long)DIM_HEAD, 64u, 32u);
    tdm_load_2d(ldsV_off, vBase, (unsigned)SEQ, 64u,
                (unsigned long long)SEQ, 32u, 64u);
  }
#endif
  for (int it = 0; it < NKB; ++it) {
    const int kb = it * 32;
    const int cur = it & 1;
#if TDM_ATTN
    if (wave == 0) {
      if (it + 1 < NKB) {  // prefetch next K/V pair into the other buffer
        tdm_load_2d(ldsK_off + (cur ^ 1) * kBytes,
                    kBase + (size_t)(kb + 32) * DIM_HEAD, 64u, (unsigned)SEQ,
                    (unsigned long long)DIM_HEAD, 64u, 32u);
        tdm_load_2d(ldsV_off + (cur ^ 1) * vBytes,
                    vBase + kb + 32, (unsigned)SEQ, 64u,
                    (unsigned long long)SEQ, 32u, 64u);
        __builtin_amdgcn_s_wait_tensorcnt((short)2);  // current pair done
      } else {
        __builtin_amdgcn_s_wait_tensorcnt((short)0);
      }
    }
#else
    {  // cooperative vector-load fallback
      int rowK = t >> 2, colK = (t & 3) * 16;
      const __bf16* kp = kBase + (size_t)(kb + rowK) * DIM_HEAD + colK;
      *(bf16x8*)&ldsK[cur][rowK][colK]     = *(const bf16x8*)kp;
      *(bf16x8*)&ldsK[cur][rowK][colK + 8] = *(const bf16x8*)(kp + 8);
      int rowV = t >> 1, colV = (t & 1) * 16;
      const __bf16* vp = vBase + (size_t)rowV * SEQ + kb + colV;
      *(bf16x8*)&ldsVT[cur][rowV][colV]     = *(const bf16x8*)vp;
      *(bf16x8*)&ldsVT[cur][rowV][colV + 8] = *(const bf16x8*)(vp + 8);
    }
#endif
    __syncthreads();

    // S = Q @ K^T : two 16-col tiles, each two chained K=32 WMMAs over dh
    v8f s[2];
    #pragma unroll
    for (int nt = 0; nt < 2; ++nt) {
      #pragma unroll
      for (int e = 0; e < 8; ++e) s[nt][e] = 0.f;
      #pragma unroll
      for (int c = 0; c < 2; ++c) {
        FragAB bfK;
        bfK.h[0] = *(const bf16x8*)&ldsK[cur][nt * 16 + rL][c * 32 + kh];
        bfK.h[1] = *(const bf16x8*)&ldsK[cur][nt * 16 + rL][c * 32 + 16 + kh];
        s[nt] = wmma_bf16(qa[c].v, bfK.v, s[nt]);
      }
    }

    // online softmax; row reductions across the 16-lane half via shfl_xor
    float p0[8], p1[8];
    #pragma unroll
    for (int i = 0; i < 8; ++i) {
      float s0 = s[0][i] * scale, s1 = s[1][i] * scale;
      float rmax = fmaxf(s0, s1);
      #pragma unroll
      for (int off = 1; off < 16; off <<= 1)
        rmax = fmaxf(rmax, __shfl_xor(rmax, off, 32));
      float mnew = fmaxf(mrow[i], rmax);
      float alpha = __expf(mrow[i] - mnew);
      mrow[i] = mnew;
      p0[i] = __expf(s0 - mnew);
      p1[i] = __expf(s1 - mnew);
      float rsum = p0[i] + p1[i];
      #pragma unroll
      for (int off = 1; off < 16; off <<= 1)
        rsum += __shfl_xor(rsum, off, 32);
      lrow[i] = lrow[i] * alpha + rsum;
      #pragma unroll
      for (int j = 0; j < 4; ++j) o[j][i] *= alpha;
    }

    // stage P through per-wave LDS (in-order DS per wave) -> A-fragment layout
    #pragma unroll
    for (int i = 0; i < 8; ++i) {
      ldsP[wave][i + hi8][rL]      = (__bf16)p0[i];
      ldsP[wave][i + hi8][16 + rL] = (__bf16)p1[i];
    }
    FragAB pa;
    pa.h[0] = *(const bf16x8*)&ldsP[wave][rL][kh];
    pa.h[1] = *(const bf16x8*)&ldsP[wave][rL][16 + kh];
    #pragma unroll
    for (int j = 0; j < 4; ++j) {
      FragAB bfV;
      bfV.h[0] = *(const bf16x8*)&ldsVT[cur][j * 16 + rL][kh];
      bfV.h[1] = *(const bf16x8*)&ldsVT[cur][j * 16 + rL][16 + kh];
      o[j] = wmma_bf16(pa.v, bfV.v, o[j]);
    }
    __syncthreads();
  }

  // epilogue: normalize, gate, merge heads
  #pragma unroll
  for (int i = 0; i < 8; ++i) {
    int n = q0row + i + hi8;
    float g = gates[(size_t)(bIdx * SEQ + n) * HEADS + h];
    float invl = 1.0f / lrow[i];
    #pragma unroll
    for (int j = 0; j < 4; ++j) {
      out[(size_t)(bIdx * SEQ + n) * (HEADS * DIM_HEAD) + h * DIM_HEAD + j * 16 + rL] =
          (__bf16)(o[j][i] * invl * g);
    }
  }
}

// ------------------------------------------------------------------ launch
extern "C" void kernel_launch(void* const* d_in, const int* in_sizes, int n_in,
                              void* d_out, int out_size, void* d_ws, size_t ws_size,
                              hipStream_t stream) {
  (void)in_sizes; (void)n_in; (void)out_size; (void)ws_size;
  const float* x       = (const float*)d_in[0];
  const float* gamma   = (const float*)d_in[1];
  const float* w_qkv   = (const float*)d_in[2];
  const float* w_gates = (const float*)d_in[3];
  const float* b_gates = (const float*)d_in[4];
  const float* w_out   = (const float*)d_in[5];
  const float* freqs   = (const float*)d_in[6];
  float* out = (float*)d_out;

  char* ws = (char*)d_ws;
  size_t off = 0;
  auto take = [&](size_t bytes) {
    char* p = ws + off;
    off = (off + bytes + 255) & ~(size_t)255;
    return p;
  };
  __bf16* xn      = (__bf16*)take((size_t)BN * DIMD * 2);
  __bf16* wqkvT   = (__bf16*)take((size_t)DIMD * QKV_N * 2);  // 3072 x 1024
  __bf16* woutT   = (__bf16*)take((size_t)DIMD * DIMD * 2);   // 1024 x 1024
  __bf16* qkv     = (__bf16*)take((size_t)BN * QKV_N * 2);
  __bf16* qd      = (__bf16*)take((size_t)BATCH * HEADS * SEQ * DIM_HEAD * 2);
  __bf16* kd      = (__bf16*)take((size_t)BATCH * HEADS * SEQ * DIM_HEAD * 2);
  __bf16* vdT     = (__bf16*)take((size_t)BATCH * HEADS * SEQ * DIM_HEAD * 2);
  float*  gts     = (float*)take((size_t)BN * HEADS * 4);
  __bf16* attn    = (__bf16*)take((size_t)BN * DIMD * 2);

  cvt_transpose_bf16<<<(DIMD * QKV_N) / 256, 256, 0, stream>>>(w_qkv, wqkvT, DIMD, QKV_N);
  cvt_transpose_bf16<<<(DIMD * DIMD) / 256, 256, 0, stream>>>(w_out, woutT, DIMD, DIMD);
  rmsnorm_gates<<<BN, 256, 0, stream>>>(x, gamma, w_gates, b_gates, xn, gts);
  gemm_bf16_wmma<__bf16><<<dim3(QKV_N / 128, BN / 128), 256, 0, stream>>>(
      xn, wqkvT, qkv, BN, QKV_N, DIMD);
  rope_scatter<<<(BN * HEADS * 32) / 256, 256, 0, stream>>>(qkv, freqs, qd, kd, vdT);
  flash_attn_gated<<<dim3(SEQ / 64, BATCH * HEADS), 128, 0, stream>>>(
      qd, kd, vdT, gts, attn);
  gemm_bf16_wmma<float><<<dim3(DIMD / 128, BN / 128), 256, 0, stream>>>(
      attn, woutT, out, BN, DIMD, DIMD);
}
